// ColBERT_62319975465254
// MI455X (gfx1250) — matrix operand order repaired and checked by reading, not compiled, not run on hardware
//
#include <hip/hip_runtime.h>
#include <hip/hip_bf16.h>

// ---------------------------------------------------------------------------
// ColBERT MaxSim scoring for MI455X (gfx1250), wave32,
// v_wmma_f32_16x16x32_bf16 + Tensor Data Mover LDS staging.
//
// ws layout:
//   [0, 196608)            : Wt  = W_tok^T as bf16, [128][768]
//   [196608, 196608+4MiB)  : reps = normalized token reps bf16, [16384][128]
//                            rows 0..8191  = query tokens (qb*256 + l)
//                            rows 8192..   = doc tokens   (db*256 + m)
// ---------------------------------------------------------------------------

typedef __attribute__((ext_vector_type(16))) __bf16 v16bf;
typedef __attribute__((ext_vector_type(8)))  __bf16 v8bf;
typedef __attribute__((ext_vector_type(8)))  float  v8f;
typedef __attribute__((ext_vector_type(4)))  unsigned int u32x4;
typedef __attribute__((ext_vector_type(8)))  unsigned int u32x8;

#define WMMA_BF16(a, b, c) \
    __builtin_amdgcn_wmma_f32_16x16x32_bf16(false, (a), false, (b), (short)0, (c), false, false)

// ---------------------------------------------------------------------------
// Kernel 1: transpose W [768,128] f32 -> Wt [128,768] bf16
// ---------------------------------------------------------------------------
__global__ void wt_kernel(const float* __restrict__ W, __bf16* __restrict__ Wt) {
    int idx = blockIdx.x * 256 + threadIdx.x;   // 0 .. 98303
    int k = idx >> 7;          // 0..767
    int n = idx & 127;         // 0..127
    Wt[n * 768 + k] = (__bf16)W[idx];
}

// ---------------------------------------------------------------------------
// Kernel 2: projection + bias + L2 normalize -> bf16 reps
// One wave computes a full 16-row x 128-col output tile (8 WMMA accumulators).
// Grid: 128 blocks x 256 threads = 1024 waves = 16384/16 row tiles.
// ---------------------------------------------------------------------------
__global__ void proj_norm_kernel(const float* __restrict__ qh,
                                 const float* __restrict__ dh,
                                 const __bf16* __restrict__ Wt,
                                 const float* __restrict__ bias,
                                 __bf16* __restrict__ reps) {
    const int wave    = (blockIdx.x * blockDim.x + threadIdx.x) >> 5; // 0..1023
    const int lane    = threadIdx.x & 31;
    const int lrow    = lane & 15;
    const int half    = lane >> 4;            // 0: lanes 0-15, 1: lanes 16-31
    const int rowbase = wave * 16;

    const float* src = (rowbase < 8192) ? (qh + (size_t)rowbase * 768)
                                        : (dh + (size_t)(rowbase - 8192) * 768);
    const float* arow = src + (size_t)lrow * 768;

    v8f c[8];
#pragma unroll
    for (int t = 0; t < 8; ++t) c[t] = (v8f)(0.0f);

    // bias for the N columns this lane's accumulator elements live on
    float bv[8];
#pragma unroll
    for (int t = 0; t < 8; ++t) bv[t] = bias[16 * t + lrow];

    for (int kb = 0; kb < 768; kb += 32) {
        // A fragment (16x32 bf16): lane<16 -> K {0..15}, lane>=16 -> K {16..31}
        const int ka = kb + half * 16;
        typedef __attribute__((ext_vector_type(8))) float v8fl;
        v8fl lo = *(const v8fl*)(arow + ka);
        v8fl hi = *(const v8fl*)(arow + ka + 8);
        v16bf a;
#pragma unroll
        for (int i = 0; i < 8; ++i) {
            a[i]     = (__bf16)lo[i];
            a[i + 8] = (__bf16)hi[i];
        }
        // B fragments from Wt (row n of Wt == column n of W): contiguous 16 bf16
#pragma unroll
        for (int t = 0; t < 8; ++t) {
            const __bf16* bp = Wt + (size_t)(16 * t + lrow) * 768 + kb + half * 16;
            v16bf b = *(const v16bf*)bp;
            c[t] = WMMA_BF16(a, b, c[t]);
        }
    }

    // bias add (same column N for all 8 row-elements of a tile)
#pragma unroll
    for (int t = 0; t < 8; ++t)
#pragma unroll
        for (int r = 0; r < 8; ++r) c[t][r] += bv[t];

    // L2 norm per row: C layout => element r of lane-half h is row (r + 8h),
    // column 16t + (lane&15). Butterfly over the 16-lane half sums columns.
    float inv[8];
#pragma unroll
    for (int r = 0; r < 8; ++r) {
        float s = 0.f;
#pragma unroll
        for (int t = 0; t < 8; ++t) s += c[t][r] * c[t][r];
        s += __shfl_xor(s, 1, 32);
        s += __shfl_xor(s, 2, 32);
        s += __shfl_xor(s, 4, 32);
        s += __shfl_xor(s, 8, 32);
        float nrm = fmaxf(sqrtf(s), 1e-12f);
        inv[r] = 1.0f / nrm;
    }

    // store normalized bf16 reps
    const int rowout = rowbase + half * 8;
#pragma unroll
    for (int r = 0; r < 8; ++r) {
        __bf16* orow = reps + (size_t)(rowout + r) * 128 + lrow;
#pragma unroll
        for (int t = 0; t < 8; ++t) orow[16 * t] = (__bf16)(c[t][r] * inv[r]);
    }
}

// ---------------------------------------------------------------------------
// Kernel 3: MaxSim scores. One block per (db, qb). The 256x128 bf16 doc block
// (64 KB) is staged into LDS by the Tensor Data Mover (one tensor_load_to_lds
// issued by wave 0, synchronized with s_wait_tensorcnt). 8 waves each own 32
// doc-token columns (2 WMMA N-tiles); per-row max -> mask -> sum.
// ---------------------------------------------------------------------------
__global__ void scores_kernel(const __bf16* __restrict__ reps,
                              const int* __restrict__ mask,
                              float* __restrict__ out) {
    const int db = blockIdx.x;
    const int qb = blockIdx.y;

    __shared__ __align__(16) __bf16 dtile[256 * 128];   // 64 KB (LDS: 320 KB/WGP)
    __shared__ float  wmax[8][16];
    __shared__ float  accs;
    __shared__ int    seps;

    const int tid  = threadIdx.x;
    const int lane = tid & 31;
    const int w    = tid >> 5;
    const int lrow = lane & 15;
    const int half = lane >> 4;

    const __bf16* qbase = reps + (size_t)(qb * 256) * 128;

    // Prefetch the query block (64 KB) toward L2/L0 while the TDM runs.
    __builtin_prefetch((const char*)qbase + tid * 256, 0, 3);

    if (tid < 32) {
        // --- Tensor DMA: stage doc block [256x128 bf16] = 65536 B into LDS ---
        // D# (CDNA5 ISA): 2D descriptor, one line of 8192 x 8B.
        unsigned long long ga =
            (unsigned long long)(const void*)(reps + (size_t)(8192 + db * 256) * 128);
        // Generic LDS pointers carry the LDS byte offset in their low 32 bits.
        unsigned lds_addr = (unsigned)(unsigned long long)(void*)dtile;

        u32x4 g0;
        g0[0] = 1u;                                   // count=1, user mode, no gather
        g0[1] = lds_addr;                             // lds_addr[31:0]
        g0[2] = (unsigned)ga;                         // global_addr[31:0]
        g0[3] = ((unsigned)(ga >> 32) & 0x1FFFFFFu)   // global_addr[56:32]
                | 0x80000000u;                        // type=2 ("image") in [127:126]

        u32x8 g1;
        g1[0] = 3u << 16;                 // data_size=3 (8B); no multicast/pad/iterate
        g1[1] = (8192u & 0xFFFFu) << 16;  // tensor_dim0[15:0]  (bits 63:48)
        g1[2] = (8192u >> 16)             // tensor_dim0[31:16] (bits 79:64)
                | (1u << 16);             // tensor_dim1[15:0] = 1 (bits 95:80)
        g1[3] = (8192u & 0xFFFFu) << 16;  // tile_dim0 = 8192   (bits 127:112)
        g1[4] = 1u;                       // tile_dim1=1, tile_dim2=0
        g1[5] = 8192u;                    // tensor_dim0_stride[31:0]
        g1[6] = (8192u & 0xFFFFu) << 16;  // stride0 hi=0 | tensor_dim1_stride[15:0]
        g1[7] = 0u;                       // tensor_dim1_stride[47:16]

        asm volatile("tensor_load_to_lds %0, %1" :: "s"(g0), "s"(g1) : "memory");

        // sep position = (sum of mask row) - 1 (overlaps with DMA)
        int s = 0;
        for (int i = lane; i < 256; i += 32) s += mask[qb * 256 + i];
        s += __shfl_xor(s, 16, 32);
        s += __shfl_xor(s, 8, 32);
        s += __shfl_xor(s, 4, 32);
        s += __shfl_xor(s, 2, 32);
        s += __shfl_xor(s, 1, 32);
        if (lane == 0) { seps = s - 1; accs = 0.f; }

        __builtin_amdgcn_s_wait_tensorcnt(0);
    }
    __syncthreads();

    for (int lt = 0; lt < 16; ++lt) {
        v8f c0 = (v8f)(0.0f), c1 = (v8f)(0.0f);
        const __bf16* aptr = qbase + (size_t)(lt * 16 + lrow) * 128;

#pragma unroll
        for (int kb = 0; kb < 128; kb += 32) {
            // A fragment from global q reps (L2 resident, 4 KB tile)
            v8bf alo = *(const v8bf*)(aptr + kb + half * 16);
            v8bf ahi = *(const v8bf*)(aptr + kb + half * 16 + 8);
            v16bf a = __builtin_shufflevector(alo, ahi,
                0, 1, 2, 3, 4, 5, 6, 7, 8, 9, 10, 11, 12, 13, 14, 15);
            // B fragments from LDS: column n == doc token row n -> contiguous K
            v16bf b0 = *(const v16bf*)&dtile[(w * 32 + lrow) * 128 + kb + half * 16];
            v16bf b1 = *(const v16bf*)&dtile[(w * 32 + 16 + lrow) * 128 + kb + half * 16];
            c0 = WMMA_BF16(a, b0, c0);
            c1 = WMMA_BF16(a, b1, c1);
        }

        // per-row max over this wave's 32 doc columns
        float mx[8];
#pragma unroll
        for (int r = 0; r < 8; ++r) mx[r] = fmaxf(c0[r], c1[r]);
#pragma unroll
        for (int r = 0; r < 8; ++r) {
            mx[r] = fmaxf(mx[r], __shfl_xor(mx[r], 1, 32));
            mx[r] = fmaxf(mx[r], __shfl_xor(mx[r], 2, 32));
            mx[r] = fmaxf(mx[r], __shfl_xor(mx[r], 4, 32));
            mx[r] = fmaxf(mx[r], __shfl_xor(mx[r], 8, 32));
        }
        if (lrow == 0) {
#pragma unroll
            for (int r = 0; r < 8; ++r) wmax[w][half * 8 + r] = mx[r];
        }
        __syncthreads();

        // combine across 8 waves, apply mask (skip CLS l=0 and SEP), accumulate
        float partial = 0.f;
        if (tid < 16) {
            float m = wmax[0][tid];
#pragma unroll
            for (int ww = 1; ww < 8; ++ww) m = fmaxf(m, wmax[ww][tid]);
            const int l = lt * 16 + tid;
            const float wgt = (l >= 1 && l != seps) ? (float)mask[qb * 256 + l] : 0.f;
            partial = m * wgt;
        }
        if (tid < 32) {
            partial += __shfl_xor(partial, 1, 32);
            partial += __shfl_xor(partial, 2, 32);
            partial += __shfl_xor(partial, 4, 32);
            partial += __shfl_xor(partial, 8, 32);
        }
        if (tid == 0) accs += partial;
        __syncthreads();
    }

    if (tid == 0) out[qb * 32 + db] = accs;
}

// ---------------------------------------------------------------------------
extern "C" void kernel_launch(void* const* d_in, const int* in_sizes, int n_in,
                              void* d_out, int out_size, void* d_ws, size_t ws_size,
                              hipStream_t stream) {
    const float* qh   = (const float*)d_in[0];   // [32,256,768]
    const float* dh   = (const float*)d_in[1];   // [32,256,768]
    const float* W    = (const float*)d_in[2];   // [768,128]
    const float* b    = (const float*)d_in[3];   // [128]
    const int*   mask = (const int*)d_in[4];     // [32,256]
    float* out = (float*)d_out;                  // [32,32]

    __bf16* Wt   = (__bf16*)d_ws;                                  // 192 KB
    __bf16* reps = (__bf16*)((char*)d_ws + (size_t)768 * 128 * 2); // 4 MB

    wt_kernel<<<384, 256, 0, stream>>>(W, Wt);
    proj_norm_kernel<<<128, 256, 0, stream>>>(qh, dh, Wt, b, reps);
    dim3 sgrid(32, 32);
    scores_kernel<<<sgrid, 256, 0, stream>>>(reps, mask, out);
}
